// Model_6150393168181
// MI455X (gfx1250) — matrix operand order repaired and checked by reading, not compiled, not run on hardware
//
#include <hip/hip_runtime.h>

typedef __attribute__((ext_vector_type(8)))  float          v8f;
typedef __attribute__((ext_vector_type(16))) __bf16         v16bf;
typedef __attribute__((ext_vector_type(8)))  unsigned short v8us;

static __device__ __forceinline__ unsigned short f2bf(float f) {
  union { float f; unsigned u; } v; v.f = f;
  unsigned r = v.u + 0x7FFFu + ((v.u >> 16) & 1u);   // round-to-nearest-even
  return (unsigned short)(r >> 16);
}
static __device__ __forceinline__ float bf2f(unsigned short h) {
  union { unsigned u; float f; } v; v.u = ((unsigned)h) << 16;
  return v.f;
}

union FragU { v8us h[2]; v16bf v; };

// ---------------------------------------------------------------------------
// C[M,N] = alpha * A[M,K] @ B[N,K]^T (+ bias[N]) (+ C_old)
// A,B bf16 row-major; C f32 or bf16.  Batched over blockIdx.z with element
// strides bsA/bsB/bsC.  Requires M%64==0, N%64==0, K%32==0, ld%16==0.
// Block = 128 threads = 4 wave32; block tile 64(M) x 64(N).
//
// Tile staging uses the CDNA5 async global->LDS engine
// (GLOBAL_LOAD_ASYNC_TO_LDS_B128, ASYNCcnt) with double buffering:
//   prologue: stage buf0
//   loop:     s_wait_asynccnt 0 ; barrier ; stage next tile into buf^1 ;
//             4x v_wmma_f32_16x16x32_bf16 from buf
// The per-wave asynccnt wait before the barrier-signal makes all waves' async
// writes visible to every wave after the barrier completes.
// ---------------------------------------------------------------------------
template<bool ADD_BIAS, bool ACCUM, bool OUT_BF16>
__global__ __launch_bounds__(128) void gemm_bf16_nt(
    int M, int N, int K,
    const unsigned short* __restrict__ A,  int lda, long long bsA,
    const unsigned short* __restrict__ Bm, int ldb, long long bsB,
    float* __restrict__ C, unsigned short* __restrict__ Cb, int ldc, long long bsC,
    const float* __restrict__ bias, float alpha)
{
  (void)M; (void)N;
  // [buf][A|B][64 rows x 32 K padded to 40] bf16 ; 80B rows spread LDS banks
  constexpr int TILE = 64 * 40;                 // ushorts per matrix tile
  __shared__ unsigned short Sm[2 * 2 * TILE];   // 20.5 KB

  const int tid  = threadIdx.x;
  const int wave = tid >> 5;
  const int lane = tid & 31;
  const int half = lane >> 4;
  const int l16  = lane & 15;
  const int bm   = blockIdx.y * 64;
  const int bn   = blockIdx.x * 64;
  const long long z = blockIdx.z;
  A  += z * bsA;
  Bm += z * bsB;

  v8f acc[4] = {v8f{0.f}, v8f{0.f}, v8f{0.f}, v8f{0.f}};

  const int lr = tid >> 1;          // 0..63: tile row staged by this thread
  const int lc = (tid & 1) * 16;    // 0 or 16: K offset staged by this thread
  const unsigned short* aPtr = A  + (long long)(bm + lr) * lda + lc;
  const unsigned short* bPtr = Bm + (long long)(bn + lr) * ldb + lc;
  // LDS byte addresses for this thread's staging slots (low 32 bits of the
  // generic pointer are the LDS offset within the workgroup allocation)
  const unsigned ldA = (unsigned)(uintptr_t)&Sm[lr * 40 + lc];
  const unsigned ldB = (unsigned)(uintptr_t)&Sm[TILE + lr * 40 + lc];

  auto stage = [&](int k0, int buf) {
    const unsigned short* ga = aPtr + k0;
    const unsigned short* gb = bPtr + k0;
    const unsigned la = ldA + (unsigned)(buf * 2 * TILE * 2);
    const unsigned lb = ldB + (unsigned)(buf * 2 * TILE * 2);
    asm volatile("global_load_async_to_lds_b128 %0, %1, off"
                 :: "v"(la),       "v"(ga)     : "memory");
    asm volatile("global_load_async_to_lds_b128 %0, %1, off"
                 :: "v"(la + 16u), "v"(ga + 8) : "memory");
    asm volatile("global_load_async_to_lds_b128 %0, %1, off"
                 :: "v"(lb),       "v"(gb)     : "memory");
    asm volatile("global_load_async_to_lds_b128 %0, %1, off"
                 :: "v"(lb + 16u), "v"(gb + 8) : "memory");
  };

  stage(0, 0);
  int buf = 0;
  for (int k0 = 0; k0 < K; k0 += 32) {
    asm volatile("s_wait_asynccnt 0x0" ::: "memory");
    __syncthreads();
    if (k0 + 32 < K) stage(k0 + 32, buf ^ 1);

    const unsigned short* As = &Sm[buf * 2 * TILE];
    const unsigned short* Bs = &Sm[buf * 2 * TILE + TILE];

    // A fragment (16x32): lane l -> row l%16; K = 8*half+{0..7}, 16+8*half+{0..7}
    FragU af;
    const int ar = wave * 16 + l16;
    af.h[0] = *(const v8us*)&As[ar * 40 + 8 * half];
    af.h[1] = *(const v8us*)&As[ar * 40 + 8 * half + 16];

#pragma unroll
    for (int nt = 0; nt < 4; ++nt) {
      // B fragment (32x16): lane l -> col l%16; K = 8*half+{0..7}, 16+8*half+{0..7}
      FragU bfr;
      const int br = nt * 16 + l16;
      bfr.h[0] = *(const v8us*)&Bs[br * 40 + 8 * half];
      bfr.h[1] = *(const v8us*)&Bs[br * 40 + 8 * half + 16];
      acc[nt] = __builtin_amdgcn_wmma_f32_16x16x32_bf16(
          false, af.v, false, bfr.v, (short)0, acc[nt], false, false);
    }
    buf ^= 1;
  }

  // epilogue: C layout: VGPR r -> row r + 8*(lane/16); col = lane%16
  const int row0 = bm + wave * 16 + half * 8;
#pragma unroll
  for (int nt = 0; nt < 4; ++nt) {
    const int col = bn + nt * 16 + l16;
    float badd = 0.f;
    if constexpr (ADD_BIAS) badd = bias[col];
#pragma unroll
    for (int r = 0; r < 8; ++r) {
      const long long idx = z * bsC + (long long)(row0 + r) * ldc + col;
      float v = acc[nt][r] * alpha + badd;
      if constexpr (OUT_BF16) {
        if constexpr (ACCUM) v += bf2f(Cb[idx]);
        Cb[idx] = f2bf(v);
      } else {
        if constexpr (ACCUM) v += C[idx];
        C[idx] = v;
      }
    }
  }
}

// --------------------------- elementwise helpers ---------------------------

__global__ void cvt_f32_bf16(const float* __restrict__ in, unsigned short* __restrict__ out, int n) {
  int i = blockIdx.x * blockDim.x + threadIdx.x;
  if (i < n) out[i] = f2bf(in[i]);
}

// rms over rows of length 128, f32 in -> bf16 out
__global__ __launch_bounds__(128) void rms128_f32(const float* __restrict__ in,
                                                  unsigned short* __restrict__ out) {
  __shared__ float s[128];
  const long long row = blockIdx.x;
  const int t = threadIdx.x;
  float x = in[row * 128 + t];
  s[t] = x * x;
  __syncthreads();
  for (int off = 64; off > 0; off >>= 1) {
    if (t < off) s[t] += s[t + off];
    __syncthreads();
  }
  float inv = rsqrtf(s[0] * (1.0f / 128.0f));
  out[row * 128 + t] = f2bf(x * inv);
}

// f = rms(relu(x)) over rows of 128, bf16 in -> bf16 out
__global__ __launch_bounds__(128) void relu_rms_bf(const unsigned short* __restrict__ in,
                                                   unsigned short* __restrict__ out) {
  __shared__ float s[128];
  const long long row = blockIdx.x;
  const int t = threadIdx.x;
  float x = fmaxf(bf2f(in[row * 128 + t]), 0.f);
  s[t] = x * x;
  __syncthreads();
  for (int off = 64; off > 0; off >>= 1) {
    if (t < off) s[t] += s[t + off];
    __syncthreads();
  }
  float inv = rsqrtf(s[0] * (1.0f / 128.0f));
  out[row * 128 + t] = f2bf(x * inv);
}

// L2-normalize rows (D = 128 or 256), f32 in -> bf16 out
__global__ __launch_bounds__(128) void l2norm_rows(const float* __restrict__ in,
                                                   unsigned short* __restrict__ out, int D) {
  __shared__ float s[128];
  const long long row = blockIdx.x;
  const int t = threadIdx.x;
  const float* r = in + row * D;
  float x0 = r[t];
  float x1 = (D > 128) ? r[128 + t] : 0.f;
  s[t] = x0 * x0 + x1 * x1;
  __syncthreads();
  for (int off = 64; off > 0; off >>= 1) {
    if (t < off) s[t] += s[t + off];
    __syncthreads();
  }
  float inv = 1.0f / fmaxf(sqrtf(s[0]), 1e-12f);
  out[row * D + t] = f2bf(x0 * inv);
  if (D > 128) out[row * D + 128 + t] = f2bf(x1 * inv);
}

// SAGE mean-aggregation: edges = triu(128,k=1) among the first 128 flattened
// rows only.  Agg[j] = (sum_{i<j} X[i]) / max(j,1); rows >=128 have Agg = 0.
__global__ void agg_kernel(const unsigned short* __restrict__ X,
                           unsigned short* __restrict__ Agg, int D) {
  int c = blockIdx.x * blockDim.x + threadIdx.x;
  if (c >= D) return;
  float s = 0.f;
  for (int j = 0; j < 128; ++j) {
    float d = (j == 0) ? 1.f : (float)j;
    Agg[j * D + c] = f2bf(s / d);
    s += bf2f(X[j * D + c]);
  }
}

// batched 128x128 transpose; optional f32->bf16 conversion on input
template<bool IN_F32>
__global__ __launch_bounds__(256) void transpose128(const void* __restrict__ in_,
                                                    unsigned short* __restrict__ out) {
  __shared__ unsigned short t[128][129];
  const long long base = (long long)blockIdx.x * 16384;
  for (int e = threadIdx.x; e < 16384; e += 256) {
    const int r = e >> 7, c = e & 127;
    unsigned short v;
    if constexpr (IN_F32) v = f2bf(((const float*)in_)[base + e]);
    else                  v = ((const unsigned short*)in_)[base + e];
    t[r][c] = v;
  }
  __syncthreads();
  for (int e = threadIdx.x; e < 16384; e += 256) {
    const int r = e >> 7, c = e & 127;
    out[base + e] = t[c][r];
  }
}

// Spack[w][p][m] = bf16(s2w[p][m*8 + w])  (extract per-worker column slice)
__global__ void spack_kernel(const float* __restrict__ s2w, unsigned short* __restrict__ sp) {
  int i = blockIdx.x * blockDim.x + threadIdx.x;
  if (i >= 8 * 128 * 128) return;
  int w = i >> 14, p = (i >> 7) & 127, m = i & 127;
  sp[i] = f2bf(s2w[p * 1024 + m * 8 + w]);
}

// ---------------------------------------------------------------------------

extern "C" void kernel_launch(void* const* d_in, const int* in_sizes, int n_in,
                              void* d_out, int out_size, void* d_ws, size_t ws_size,
                              hipStream_t stream) {
  (void)in_sizes; (void)n_in; (void)out_size; (void)ws_size;
  const int Rows = 256 * 128;                 // B*N = 32768 flattened node rows

  const float* flows = (const float*)d_in[0];
  const float* g1_wl = (const float*)d_in[10];
  const float* g1_bl = (const float*)d_in[11];
  const float* g1_wr = (const float*)d_in[12];
  const float* g2_wl = (const float*)d_in[13];
  const float* g2_bl = (const float*)d_in[14];
  const float* g2_wr = (const float*)d_in[15];
  const float* g3_wl = (const float*)d_in[16];
  const float* g3_bl = (const float*)d_in[17];
  const float* g3_wr = (const float*)d_in[18];
  const float* g4_wl = (const float*)d_in[19];
  const float* g4_bl = (const float*)d_in[20];
  const float* g4_wr = (const float*)d_in[21];
  const float* qsa   = (const float*)d_in[24];
  const float* ksa   = (const float*)d_in[25];
  const float* s2w   = (const float*)d_in[27];
  float* outp = (float*)d_out;

  // ---- workspace carve (all offsets 256B aligned) ----
  char* ws = (char*)d_ws;
  size_t off = 0;
  auto carve = [&](size_t bytes) -> char* {
    char* p = ws + off;
    off += (bytes + 255) & ~(size_t)255;
    return p;
  };
  unsigned short* Xa   = (unsigned short*)carve((size_t)Rows * 256 * 2);  // 16MB
  unsigned short* Xb   = (unsigned short*)carve((size_t)Rows * 256 * 2);  // 16MB
  float*          Yf   = (float*)carve((size_t)Rows * 256 * 4);           // 32MB
  unsigned short* Agg  = (unsigned short*)carve((size_t)128 * 256 * 2);
  unsigned short* w1l  = (unsigned short*)carve(16384 * 2);
  unsigned short* w1r  = (unsigned short*)carve(16384 * 2);
  unsigned short* w2l  = (unsigned short*)carve(32768 * 2);
  unsigned short* w2r  = (unsigned short*)carve(32768 * 2);
  unsigned short* w3l  = (unsigned short*)carve(32768 * 2);
  unsigned short* w3r  = (unsigned short*)carve(32768 * 2);
  unsigned short* w4l  = (unsigned short*)carve(16384 * 2);
  unsigned short* w4r  = (unsigned short*)carve(16384 * 2);
  unsigned short* qT   = (unsigned short*)carve((size_t)8 * 16384 * 2);
  unsigned short* kT   = (unsigned short*)carve((size_t)8 * 16384 * 2);
  unsigned short* Mtbf = (unsigned short*)carve((size_t)8 * 16384 * 2);
  unsigned short* Spk  = (unsigned short*)carve((size_t)8 * 16384 * 2);
  unsigned short* fT   = (unsigned short*)carve((size_t)256 * 16384 * 2); // 8MB
  unsigned short* Gm   = (unsigned short*)Yf;   // alias: Yf is dead in worker phase
  unsigned short* Hm   = Xa;                    // alias: Xa is dead after relu_rms

  // ---- weight bf16 packing ----
  auto cvt = [&](const float* src, unsigned short* dst, int n) {
    cvt_f32_bf16<<<(n + 255) / 256, 256, 0, stream>>>(src, dst, n);
  };
  cvt(g1_wl, w1l, 16384); cvt(g1_wr, w1r, 16384);
  cvt(g2_wl, w2l, 32768); cvt(g2_wr, w2r, 32768);
  cvt(g3_wl, w3l, 32768); cvt(g3_wr, w3r, 32768);
  cvt(g4_wl, w4l, 16384); cvt(g4_wr, w4r, 16384);

  // ---- X0 = rms(flows) over last dim ----
  rms128_f32<<<Rows, 128, 0, stream>>>(flows, Xa);

  // ---- 4 SAGE layers: Y = X@Wr^T + bl (+ Agg@Wl^T on rows 0..127), L2-norm ----
  struct Layer { const unsigned short* wl; const float* bl; const unsigned short* wr; int Din, Dout; };
  Layer Ls[4] = {
    {w1l, g1_bl, w1r, 128, 128},
    {w2l, g2_bl, w2r, 128, 256},
    {w3l, g3_bl, w3r, 256, 128},
    {w4l, g4_bl, w4r, 128, 128},
  };
  unsigned short* cur = Xa;
  unsigned short* nxt = Xb;
  for (int li = 0; li < 4; ++li) {
    const Layer& L = Ls[li];
    agg_kernel<<<(L.Din + 127) / 128, 128, 0, stream>>>(cur, Agg, L.Din);
    gemm_bf16_nt<true, false, false><<<dim3(L.Dout / 64, Rows / 64, 1), 128, 0, stream>>>(
        Rows, L.Dout, L.Din, cur, L.Din, 0, L.wr, L.Din, 0,
        Yf, nullptr, L.Dout, 0, L.bl, 1.0f);
    gemm_bf16_nt<false, true, false><<<dim3(L.Dout / 64, 2, 1), 128, 0, stream>>>(
        128, L.Dout, L.Din, Agg, L.Din, 0, L.wl, L.Din, 0,
        Yf, nullptr, L.Dout, 0, nullptr, 1.0f);
    l2norm_rows<<<Rows, 128, 0, stream>>>(Yf, nxt, L.Dout);
    unsigned short* t = cur; cur = nxt; nxt = t;
  }
  // cur == Xa holds layer-4 output; f = rms(relu(x)) -> Xb
  relu_rms_bf<<<Rows, 128, 0, stream>>>(cur, nxt);
  unsigned short* fbf = nxt;   // == Xb

  // ---- Worker (self-attention branch only; ca branch is dead code) ----
  // sa[b,w] = f_b (qsa_w^T ksa_w / sqrt(128)) f_b^T ; out_b = sum_w sa[b,w] S_w^T
  transpose128<true ><<<8,   256, 0, stream>>>((const void*)qsa, qT);   // qT[w][d][o]
  transpose128<true ><<<8,   256, 0, stream>>>((const void*)ksa, kT);   // kT[w][d'][o]
  transpose128<false><<<256, 256, 0, stream>>>((const void*)fbf, fT);   // fT[b][d][n]
  spack_kernel<<<(8 * 16384 + 255) / 256, 256, 0, stream>>>(s2w, Spk);  // S_w[p][m]

  // Mt_w = kT_w @ qT_w^T / sqrt(128)   (batched over w)
  const float inv_scale = 0.08838834764831845f;   // 1/sqrt(128)
  gemm_bf16_nt<false, false, true><<<dim3(2, 2, 8), 128, 0, stream>>>(
      128, 128, 128, kT, 128, 16384, qT, 128, 16384,
      nullptr, Mtbf, 128, 16384, nullptr, inv_scale);

  for (int w = 0; w < 8; ++w) {
    // G[b] = f_b @ Mt_w^T        (batched over b=256)
    gemm_bf16_nt<false, false, true><<<dim3(2, 2, 256), 128, 0, stream>>>(
        128, 128, 128, fbf, 128, 16384, Mtbf + (long long)w * 16384, 128, 0,
        nullptr, Gm, 128, 16384, nullptr, 1.0f);
    // H[b] = S_w @ f_b  ==  S_w @ (fT_b)^T
    gemm_bf16_nt<false, false, true><<<dim3(2, 2, 256), 128, 0, stream>>>(
        128, 128, 128, Spk + (long long)w * 16384, 128, 0, fT, 128, 16384,
        nullptr, Hm, 128, 16384, nullptr, 1.0f);
    // out_b += G[b] @ H[b]^T
    if (w == 0)
      gemm_bf16_nt<false, false, false><<<dim3(2, 2, 256), 128, 0, stream>>>(
          128, 128, 128, Gm, 128, 16384, Hm, 128, 16384,
          outp, nullptr, 128, 16384, nullptr, 1.0f);
    else
      gemm_bf16_nt<false, true, false><<<dim3(2, 2, 256), 128, 0, stream>>>(
          128, 128, 128, Gm, 128, 16384, Hm, 128, 16384,
          outp, nullptr, 128, 16384, nullptr, 1.0f);
  }
}
